// BiLSTM_CRF_7172595384723
// MI455X (gfx1250) — compile-verified, matrix operand order checked by reading, not run on hardware
//
#include <hip/hip_runtime.h>
#include <hip/hip_bf16.h>
#include <stddef.h>

// ---------------------------------------------------------------------------
// BiLSTM-CRF for MI455X (gfx1250, wave32, WMMA).
//
// Pipeline:
//   K0  packbf      : f32 weights -> bf16 (K/N padded) in workspace
//   K1  embed       : gather emb_table rows -> bf16 X [32768 x 320]
//   K2  gemm_in     : X @ Wih^T + (bih+bhh)  (both dirs) -> preact f32,
//                     stored time-major [(t*64+b) x 1024] for the recurrence
//   K3  lstm        : 2 blocks (fwd/bwd), 32 waves each; h(bf16),c(f32),
//                     gates(bf16) resident in 224KB LDS; per step:
//                     WMMA h@Whh^T (+preact as C-init), barrier, gate math.
//                     h-fragments register-cached across a wave's 8 N-tiles;
//                     next step's preact rows prefetched (global_prefetch).
//   K4  gemm_out    : lstm_out(bf16) @ W_out^T + b_out -> feats f32 (N pad 32)
//   K5  viterbi_fwd : wave-per-batch CRF forward max-product, bp bytes
//   K6  viterbi_back: scalar backtrace, writes int32 path * mask to d_out
// ---------------------------------------------------------------------------

#define Bn   64
#define Tn   512
#define En   300
#define EP   320       // E padded to multiple of 32
#define Hn   256
#define G4   1024      // 4*H
#define H2n  512
#define L2n  22
#define NT   32768     // B*T
#define NPAD 32        // L2 padded for WMMA N tiles

typedef __attribute__((ext_vector_type(16))) __bf16 v16bf;
typedef __attribute__((ext_vector_type(8)))  __bf16 v8bf;
typedef __attribute__((ext_vector_type(8)))  float  v8f;

__device__ inline __bf16 f2bf(float x) {
  unsigned u = __builtin_bit_cast(unsigned, x);
  unsigned r = (u + 0x7FFFu + ((u >> 16) & 1u)) >> 16;
  unsigned short s = (unsigned short)r;
  return __builtin_bit_cast(__bf16, s);
}
__device__ inline float bf2f(__bf16 b) {
  unsigned short s = __builtin_bit_cast(unsigned short, b);
  unsigned u = ((unsigned)s) << 16;
  return __builtin_bit_cast(float, u);
}
__device__ inline float sigm(float x) { return 1.0f / (1.0f + __expf(-x)); }

// Load one 16x32 bf16 WMMA operand fragment (A or B^T; both are row-major
// [16 rows x 32 k] slices here). Per the CDNA5 ISA 16-bit A layout:
//   lanes 0-15 : k = k0+0..7 (lo) and k0+16..23 (hi)
//   lanes16-31 : k = k0+8..15 (lo) and k0+24..31 (hi)
// p must point at  base + row*ld + k0 + 8*(lane>>4).
__device__ inline v16bf ldfrag(const __bf16* p) {
  union { v16bf v; struct { v8bf lo; v8bf hi; } s; } u;
  u.s.lo = *(const v8bf*)(p);
  u.s.hi = *(const v8bf*)(p + 16);
  return u.v;
}

// ---------------- K0: pad/convert f32 weight -> bf16 ----------------------
__global__ void packbf(const float* __restrict__ src, __bf16* __restrict__ dst,
                       int dstRows, int srcRows, int cols, int colsPad) {
  int idx = blockIdx.x * blockDim.x + threadIdx.x;
  int total = dstRows * colsPad;
  if (idx >= total) return;
  int r = idx / colsPad, c = idx % colsPad;
  float v = (r < srcRows && c < cols) ? src[r * cols + c] : 0.0f;
  dst[idx] = f2bf(v);
}

// ---------------- K1: embedding gather -> bf16, pad E to 320 --------------
__global__ void embed_k(const int* __restrict__ words,
                        const float* __restrict__ emb,
                        __bf16* __restrict__ Xg) {
  int row = blockIdx.x;          // row = b*T + t  (flat batch_word order)
  int e   = threadIdx.x;         // 320 threads
  int w   = words[row];
  float v = (e < En) ? emb[(size_t)w * En + e] : 0.0f;
  Xg[(size_t)row * EP + e] = f2bf(v);
}

// ---------------- K2: preact = X @ Wih^T + bih + bhh (both dirs) ----------
// One wave per 16x16 output tile; 10 k-steps of v_wmma_f32_16x16x32_bf16.
// Output written time-major: preact[dir][(t*64+b)*1024 + n].
__global__ void gemm_in_k(const __bf16* __restrict__ Xg,
                          const __bf16* __restrict__ Wihg,
                          const float* __restrict__ bih_f, const float* __restrict__ bhh_f,
                          const float* __restrict__ bih_b, const float* __restrict__ bhh_b,
                          float* __restrict__ preact) {
  int id   = (blockIdx.x * blockDim.x + threadIdx.x) >> 5;   // global wave id
  int lane = threadIdx.x & 31;
  int l16  = lane & 15, hiH = lane >> 4;

  int dir = id >> 17;                 // 2048*64 = 131072 tiles per dir
  int rem = id & 131071;
  int mt  = rem >> 6;                 // 2048 M tiles
  int nt  = rem & 63;                 // 64 N tiles

  const float* bih = dir ? bih_b : bih_f;
  const float* bhh = dir ? bhh_b : bhh_f;
  const __bf16* Wm = Wihg + (size_t)dir * G4 * EP;

  int n = nt * 16 + l16;
  float bias = bih[n] + bhh[n];
  v8f c;
#pragma unroll
  for (int r = 0; r < 8; ++r) c[r] = bias;

  const __bf16* arow = Xg + (size_t)(mt * 16 + l16) * EP + 8 * hiH;
  const __bf16* brow = Wm + (size_t)n * EP + 8 * hiH;
#pragma unroll
  for (int k0 = 0; k0 < EP; k0 += 32) {
    v16bf a = ldfrag(arow + k0);
    v16bf b = ldfrag(brow + k0);
    c = __builtin_amdgcn_wmma_f32_16x16x32_bf16(false, a, false, b,
                                                (short)0, c, false, false);
  }
#pragma unroll
  for (int r = 0; r < 8; ++r) {
    int mrow = mt * 16 + r + 8 * hiH;        // row in b*T+t order
    int bb = mrow >> 9, tt = mrow & (Tn - 1);
    preact[((size_t)dir * NT + (size_t)tt * Bn + bb) * G4 + n] = c[r];
  }
}

// ---------------- K3: recurrent LSTM (one block per direction) ------------
// LDS: h bf16 [64x256] 32KB | gates bf16 [64x1024] 128KB | c f32 [64x256] 64KB
__global__ __launch_bounds__(1024) void lstm_k(const float* __restrict__ preact,
                                               const __bf16* __restrict__ Whhg,
                                               const unsigned char* __restrict__ maskB,
                                               __bf16* __restrict__ lstm_out) {
  extern __shared__ char smem[];
  __bf16* hS = (__bf16*)smem;                         // 64*256
  __bf16* gS = (__bf16*)(smem + 64 * Hn * 2);         // 64*1024
  float*  cS = (float*)(smem + 64 * Hn * 2 + 64 * G4 * 2);

  int dir  = blockIdx.x;
  int tid  = threadIdx.x;
  int wave = tid >> 5, lane = tid & 31;
  int l16  = lane & 15, hiH = lane >> 4;

  for (int p = tid; p < Bn * Hn; p += 1024) { hS[p] = f2bf(0.0f); cS[p] = 0.0f; }
  __syncthreads();

  const float*  pre = preact + (size_t)dir * NT * G4;
  const __bf16* Whh = Whhg + (size_t)dir * G4 * Hn;

  // A wave's 8 tiles (tile = wave*8+i) share mt = wave>>3, nt = (wave&7)*8+i.
  int mt     = wave >> 3;
  int ntBase = (wave & 7) * 8;
  const __bf16* arow = hS + (mt * 16 + l16) * Hn + 8 * hiH;
  ptrdiff_t tstep = (ptrdiff_t)Bn * G4;     // +1 step for fwd, -1 for bwd
  if (dir) tstep = -tstep;

  for (int s = 0; s < Tn; ++s) {
    int t = (dir == 0) ? s : (Tn - 1 - s);
    // ---- phase 1: gates = preact[t] + h @ Whh^T  (32 waves x 8 tiles) ----
    // Register-cache the h fragments: identical for all 8 tiles of this wave.
    v16bf aF[8];
#pragma unroll
    for (int kk = 0; kk < 8; ++kk) aF[kk] = ldfrag(arow + kk * 32);

#pragma unroll 1
    for (int i = 0; i < 8; ++i) {
      int nt = ntBase + i;
      int n = nt * 16 + l16;
      const float* prow = pre + ((size_t)t * Bn + mt * 16 + 8 * hiH) * G4 + n;
      // prefetch next step's preact rows while we compute this one
      __builtin_prefetch(prow + tstep, 0, 1);
      v8f c;
#pragma unroll
      for (int r = 0; r < 8; ++r) c[r] = prow[(size_t)r * G4];
      const __bf16* brow = Whh + (size_t)n * Hn + 8 * hiH;
#pragma unroll
      for (int kk = 0; kk < 8; ++kk) {
        v16bf b = ldfrag(brow + kk * 32);
        c = __builtin_amdgcn_wmma_f32_16x16x32_bf16(false, aF[kk], false, b,
                                                    (short)0, c, false, false);
      }
#pragma unroll
      for (int r = 0; r < 8; ++r)
        gS[(mt * 16 + r + 8 * hiH) * G4 + n] = f2bf(c[r]);
    }
    __syncthreads();
    // ---- phase 2: elementwise gate math, update h,c, emit output ----
    for (int p = tid; p < Bn * Hn; p += 1024) {
      int bb = p >> 8, j = p & (Hn - 1);
      float gi = bf2f(gS[bb * G4 + j]);
      float gf = bf2f(gS[bb * G4 + Hn + j]);
      float gg = bf2f(gS[bb * G4 + 2 * Hn + j]);
      float go = bf2f(gS[bb * G4 + 3 * Hn + j]);
      float co = cS[p];
      float cn = sigm(gf) * co + sigm(gi) * tanhf(gg);
      float hn = sigm(go) * tanhf(cn);
      bool  m  = maskB[bb * Tn + t] != 0;
      float ho = bf2f(hS[p]);
      cS[p] = m ? cn : co;
      hS[p] = f2bf(m ? hn : ho);
      lstm_out[((size_t)t * Bn + bb) * H2n + dir * Hn + j] = f2bf(m ? hn : 0.0f);
    }
    __syncthreads();
  }
}

// ---------------- K4: feats = lstm_out @ W_out^T + b_out ------------------
__global__ void gemm_out_k(const __bf16* __restrict__ lo,
                           const __bf16* __restrict__ Woutg,
                           const float* __restrict__ b_out,
                           float* __restrict__ feats) {
  int id   = (blockIdx.x * blockDim.x + threadIdx.x) >> 5;
  int lane = threadIdx.x & 31;
  int l16  = lane & 15, hiH = lane >> 4;
  int mt = id >> 1, nt = id & 1;            // 2048 x 2 tiles
  int n = nt * 16 + l16;
  float bias = (n < L2n) ? b_out[n] : 0.0f;
  v8f c;
#pragma unroll
  for (int r = 0; r < 8; ++r) c[r] = bias;
  const __bf16* arow = lo + (size_t)(mt * 16 + l16) * H2n + 8 * hiH;
  const __bf16* brow = Woutg + (size_t)n * H2n + 8 * hiH;
#pragma unroll
  for (int k0 = 0; k0 < H2n; k0 += 32) {
    v16bf a = ldfrag(arow + k0);
    v16bf b = ldfrag(brow + k0);
    c = __builtin_amdgcn_wmma_f32_16x16x32_bf16(false, a, false, b,
                                                (short)0, c, false, false);
  }
#pragma unroll
  for (int r = 0; r < 8; ++r)
    feats[(size_t)(mt * 16 + r + 8 * hiH) * NPAD + n] = c[r];
}

// ---------------- K5: CRF Viterbi forward (wave per batch row) ------------
__global__ void viterbi_fwd_k(const float* __restrict__ feats,
                              const float* __restrict__ trans,
                              const unsigned char* __restrict__ maskB,
                              unsigned char* __restrict__ bp,
                              int* __restrict__ best_tag) {
  int gw   = (blockIdx.x * blockDim.x + threadIdx.x) >> 5;  // batch row
  int lane = threadIdx.x & 31;
  int b = gw;
  int j = lane;
  bool valid = j < L2n;
  int js = valid ? j : 0;
  const float NEG = -3.0e38f;

  float tr[L2n];
#pragma unroll
  for (int i = 0; i < L2n; ++i) tr[i] = trans[i * L2n + js];

  // part = feats[0] + trans[START=20][:]
  float part = valid ? (feats[(size_t)b * NPAD + j] + trans[20 * L2n + js]) : NEG;

  for (int t = 1; t < Tn; ++t) {
    float fj = valid ? feats[((size_t)t * Bn + b) * NPAD + j] : 0.0f;
    float best = NEG; int bi = 0;
#pragma unroll
    for (int i = 0; i < L2n; ++i) {
      float v = __shfl(part, i, 32) + tr[i];
      if (v > best) { best = v; bi = i; }
    }
    bool m = maskB[b * Tn + t] != 0;
    float pn = best + fj;
    int bpj = m ? bi : j;
    part = (m && valid) ? pn : part;
    bp[((size_t)t * Bn + b) * NPAD + j] = (unsigned char)bpj;
  }
  float fin = valid ? (part + trans[js * L2n + 21]) : NEG;   // + trans[:,STOP]
  float bv = fin; int bj = j;
#pragma unroll
  for (int off = 16; off; off >>= 1) {
    float ov = __shfl_xor(bv, off, 32);
    int   oj = __shfl_xor(bj, off, 32);
    if (ov > bv || (ov == bv && oj < bj)) { bv = ov; bj = oj; }
  }
  if (lane == 0) best_tag[b] = bj;
}

// ---------------- K6: backtrace + mask -> int32 output --------------------
__global__ void viterbi_back_k(const unsigned char* __restrict__ bp,
                               const int* __restrict__ best_tag,
                               const unsigned char* __restrict__ maskB,
                               int* __restrict__ out) {
  int b = threadIdx.x;
  if (b >= Bn) return;
  int tag = best_tag[b];
  out[b * Tn + (Tn - 1)] = maskB[b * Tn + (Tn - 1)] ? tag : 0;
  for (int t = Tn - 1; t >= 1; --t) {
    tag = bp[((size_t)t * Bn + b) * NPAD + tag];
    out[b * Tn + (t - 1)] = maskB[b * Tn + (t - 1)] ? tag : 0;
  }
}

// ---------------------------------------------------------------------------
extern "C" void kernel_launch(void* const* d_in, const int* in_sizes, int n_in,
                              void* d_out, int out_size, void* d_ws, size_t ws_size,
                              hipStream_t stream) {
  const int*   batch_word = (const int*)d_in[0];
  const unsigned char* maskB = (const unsigned char*)d_in[1];  // jnp bool = 1 byte
  const float* emb   = (const float*)d_in[4];
  const float* Wih_f = (const float*)d_in[5];
  const float* Whh_f = (const float*)d_in[6];
  const float* bih_f = (const float*)d_in[7];
  const float* bhh_f = (const float*)d_in[8];
  const float* Wih_b = (const float*)d_in[9];
  const float* Whh_b = (const float*)d_in[10];
  const float* bih_b = (const float*)d_in[11];
  const float* bhh_b = (const float*)d_in[12];
  const float* W_out = (const float*)d_in[13];
  const float* b_out = (const float*)d_in[14];
  const float* trans = (const float*)d_in[15];
  int* out = (int*)d_out;

  // Workspace layout (bytes), ~331 MB total.
  char* ws = (char*)d_ws;
  size_t off = 0;
  __bf16* Xg    = (__bf16*)(ws + off); off += (size_t)NT * EP * 2;          // 20.97 MB
  __bf16* Wihg  = (__bf16*)(ws + off); off += (size_t)2 * G4 * EP * 2;      //  1.31 MB
  __bf16* Whhg  = (__bf16*)(ws + off); off += (size_t)2 * G4 * Hn * 2;      //  1.05 MB
  __bf16* Woutg = (__bf16*)(ws + off); off += (size_t)NPAD * H2n * 2;       //  0.03 MB
  float*  preact= (float*)(ws + off);  off += (size_t)2 * NT * G4 * 4;      // 268.4 MB
  __bf16* lstmo = (__bf16*)(ws + off); off += (size_t)NT * H2n * 2;         // 33.55 MB
  float*  feats = (float*)(ws + off);  off += (size_t)NT * NPAD * 4;        //  4.19 MB
  unsigned char* bp = (unsigned char*)(ws + off); off += (size_t)Tn*Bn*NPAD;//  1.05 MB
  int* best_tag = (int*)(ws + off);    off += Bn * 4;

  // K0: weight conversion/padding
  {
    int tot = G4 * EP;
    packbf<<<(tot + 255) / 256, 256, 0, stream>>>(Wih_f, Wihg,              G4, G4, En, EP);
    packbf<<<(tot + 255) / 256, 256, 0, stream>>>(Wih_b, Wihg + (size_t)G4*EP, G4, G4, En, EP);
    tot = G4 * Hn;
    packbf<<<(tot + 255) / 256, 256, 0, stream>>>(Whh_f, Whhg,              G4, G4, Hn, Hn);
    packbf<<<(tot + 255) / 256, 256, 0, stream>>>(Whh_b, Whhg + (size_t)G4*Hn, G4, G4, Hn, Hn);
    tot = NPAD * H2n;
    packbf<<<(tot + 255) / 256, 256, 0, stream>>>(W_out, Woutg, NPAD, L2n, H2n, H2n);
  }
  // K1: embedding gather (one block per (b,t) row)
  embed_k<<<NT, EP, 0, stream>>>(batch_word, emb, Xg);
  // K2: input GEMM, 262144 tiles, 8 waves/block
  gemm_in_k<<<32768, 256, 0, stream>>>(Xg, Wihg, bih_f, bhh_f, bih_b, bhh_b, preact);
  // K3: recurrence, 1 block per direction, 224KB dynamic LDS
  {
    size_t lds = (size_t)Bn * Hn * 2 + (size_t)Bn * G4 * 2 + (size_t)Bn * Hn * 4;
    lstm_k<<<2, 1024, lds, stream>>>(preact, Whhg, maskB, lstmo);
  }
  // K4: projection, 4096 tiles, 8 waves/block
  gemm_out_k<<<512, 256, 0, stream>>>(lstmo, Woutg, b_out, feats);
  // K5: Viterbi forward, 64 waves
  viterbi_fwd_k<<<8, 256, 0, stream>>>(feats, trans, maskB, bp, best_tag);
  // K6: backtrace
  viterbi_back_k<<<1, 64, 0, stream>>>(bp, best_tag, maskB, out);
}
